// TransformerEncoder_23330262352426
// MI455X (gfx1250) — compile-verified, matrix operand order checked by reading, not run on hardware
//
#include <hip/hip_runtime.h>

// ---------------------------------------------------------------------------
// Transformer encoder for MI455X (gfx1250): bf16 WMMA everywhere GEMM-shaped.
// S=1024, B=4, E=1024, H=16, hd=64, L=4.
// GEMM W-tiles staged through LDS via gfx1250 async LDS-DMA (double buffered).
// ---------------------------------------------------------------------------

typedef __attribute__((ext_vector_type(16))) __bf16 v16bf;
typedef __attribute__((ext_vector_type(8)))  __bf16 v8bf;
typedef __attribute__((ext_vector_type(8)))  float  v8f;
typedef __attribute__((ext_vector_type(4)))  int    v4i;

#define S_  1024
#define B_  4
#define E_  1024
#define H_  16
#define HD_ 64
#define M_  (S_*B_)   // 4096 rows
#define L_  4

#if defined(__has_builtin)
#if __has_builtin(__builtin_amdgcn_global_load_async_to_lds_b128)
#define HAVE_ASYNC_LDS 1
#endif
#endif
#ifndef HAVE_ASYNC_LDS
#define HAVE_ASYNC_LDS 0
#endif

// global addrspace(1) / LDS addrspace(3) v4i pointer casts for the builtin
#define GAS1(p) ((__attribute__((address_space(1))) v4i*)(v4i*)(p))
#define LAS3(p) ((__attribute__((address_space(3))) v4i*)(v4i*)(p))

static __device__ inline v16bf pack16(v8bf lo, v8bf hi) {
    v16bf r;
#pragma unroll
    for (int i = 0; i < 8; ++i) { r[i] = lo[i]; r[i + 8] = hi[i]; }
    return r;
}

static __device__ inline v8f wmma_bf16(v16bf a, v16bf b, v8f c) {
    return __builtin_amdgcn_wmma_f32_16x16x32_bf16(
        /*neg_a=*/false, a, /*neg_b=*/false, b,
        /*c_mod=*/(short)0, c, /*reuse_a=*/false, /*reuse_b=*/false);
}

static __device__ inline void wait_async0() {
#if HAVE_ASYNC_LDS
#if __has_builtin(__builtin_amdgcn_s_wait_asynccnt)
    __builtin_amdgcn_s_wait_asynccnt(0);
#else
    asm volatile("s_wait_asynccnt 0x0" ::: "memory");
#endif
#endif
}

// ---------------------------------------------------------------------------
// fp32 -> bf16 conversion (weights)
// ---------------------------------------------------------------------------
__global__ void cvt_bf16_kernel(const float* __restrict__ in,
                                __bf16* __restrict__ out, int n) {
    int i = blockIdx.x * 256 + threadIdx.x;
    if (i < n) out[i] = (__bf16)in[i];
}

// ---------------------------------------------------------------------------
// x + positional encoding  (S,B,E) fp32
// ---------------------------------------------------------------------------
__global__ void posadd_kernel(const float* __restrict__ xin,
                              float* __restrict__ xout) {
    int i = blockIdx.x * 256 + threadIdx.x;          // < 4194304
    int e  = i & (E_ - 1);
    int sb = i >> 10;
    int s  = sb >> 2;                                 // B_ == 4
    int pairi = e >> 1;
    float div = __expf((float)(2 * pairi) * (9.210340371976184f / (float)E_));
    float ang = (float)s / div;
    float pe  = (e & 1) ? __cosf(ang) : __sinf(ang);
    xout[i] = xin[i] + pe;
}

// ---------------------------------------------------------------------------
// LayerNorm over last dim (E=1024). One block (256 thr) per row.
// ---------------------------------------------------------------------------
template <int OBF>
__global__ __launch_bounds__(256) void ln_kernel(const float* __restrict__ x,
                                                 const float* __restrict__ g,
                                                 const float* __restrict__ be,
                                                 __bf16* __restrict__ obf,
                                                 float*  __restrict__ of32) {
    int row = blockIdx.x, tid = threadIdx.x;
    int lane = tid & 31, wid = tid >> 5;
    const float* xr = x + (size_t)row * E_;

    float v[4];
#pragma unroll
    for (int i = 0; i < 4; ++i) v[i] = xr[tid + i * 256];

    __shared__ float red[8];
    __shared__ float bc0, bc1;

    float s = v[0] + v[1] + v[2] + v[3];
#pragma unroll
    for (int m = 16; m > 0; m >>= 1) s += __shfl_xor(s, m, 32);
    if (lane == 0) red[wid] = s;
    __syncthreads();
    if (tid == 0) {
        float t = 0.f;
        for (int j = 0; j < 8; ++j) t += red[j];
        bc0 = t * (1.0f / (float)E_);
    }
    __syncthreads();
    float mu = bc0;

    float q = 0.f;
#pragma unroll
    for (int i = 0; i < 4; ++i) { float d = v[i] - mu; q += d * d; }
#pragma unroll
    for (int m = 16; m > 0; m >>= 1) q += __shfl_xor(q, m, 32);
    if (lane == 0) red[wid] = q;
    __syncthreads();
    if (tid == 0) {
        float t = 0.f;
        for (int j = 0; j < 8; ++j) t += red[j];
        bc1 = rsqrtf(t * (1.0f / (float)E_) + 1e-5f);
    }
    __syncthreads();
    float rs = bc1;

#pragma unroll
    for (int i = 0; i < 4; ++i) {
        int c = tid + i * 256;
        float o = (v[i] - mu) * rs * g[c] + be[c];
        if (OBF) obf[(size_t)row * E_ + c] = (__bf16)o;
        else     of32[(size_t)row * E_ + c] = o;
    }
}

// ---------------------------------------------------------------------------
// bf16 WMMA GEMM:  C(M=4096, N) = A(M,K=1024) * W(N,K)^T  (+bias, epilogue)
// Block: 256 thr = 8 waves. Wave w -> rows [by*128+w*16, +16), cols [bx*64, +64).
// W tile (64 rows x 32 k) staged in LDS once per workgroup, double-buffered
// via async LDS-DMA (ASYNCcnt) when available.
// MODE 0: QKV scatter to q(scaled)/k/vT head buffers (N=3072)
// MODE 1: x += acc + bias (residual, fp32, N=1024)
// MODE 2: obf = bf16(relu(acc + bias)) (N=1024)
// ---------------------------------------------------------------------------
#define WROW 40   // padded LDS row stride in bf16 elems (80B: 16B-aligned, no bank dup)

template <int MODE>
__global__ __launch_bounds__(256) void gemm_bf16_kernel(
    const __bf16* __restrict__ A, const __bf16* __restrict__ W,
    const float* __restrict__ bias, float* __restrict__ xio,
    __bf16* __restrict__ obf, __bf16* __restrict__ qb,
    __bf16* __restrict__ kb, __bf16* __restrict__ vtb) {
    const int K = 1024;
    __shared__ __bf16 wt[2][64 * WROW];          // 2 x 5 KB

    int tid = threadIdx.x;
    int lane = tid & 31, wid = tid >> 5;
    int half = lane >> 4, r = lane & 15;
    int m0 = blockIdx.y * 128 + wid * 16;
    int n0 = blockIdx.x * 64;

    // staging: thread t copies 16B; row = t/4 (0..63), chunk = t%4
    int srow = tid >> 2, schunk = tid & 3;
    const __bf16* wsrc = W + (size_t)(n0 + srow) * K + schunk * 8;
    __bf16* wdst0 = &wt[0][srow * WROW + schunk * 8];
    __bf16* wdst1 = &wt[1][srow * WROW + schunk * 8];

    v8f acc[4];
#pragma unroll
    for (int nt = 0; nt < 4; ++nt)
#pragma unroll
        for (int j = 0; j < 8; ++j) acc[nt][j] = 0.f;

    const __bf16* pa_base = A + (size_t)(m0 + r) * K + half * 8;

    // prologue: stage k-step 0 into buffer 0
#if HAVE_ASYNC_LDS
    __builtin_amdgcn_global_load_async_to_lds_b128(GAS1(wsrc), LAS3(wdst0), 0, 0);
#else
    *(v8bf*)wdst0 = *(const v8bf*)wsrc;
#endif

    for (int i = 0; i < 32; ++i) {
        int k0 = i * 32;
        const __bf16* cur = (i & 1) ? wt[1] : wt[0];
        __bf16* nxt = (i & 1) ? wdst0 : wdst1;

        wait_async0();          // batch i landed (this wave's DMA)
        __syncthreads();        // all waves' batch i visible; prev reads of nxt done

        if (i + 1 < 32) {
            const __bf16* src = wsrc + k0 + 32;
#if HAVE_ASYNC_LDS
            __builtin_amdgcn_global_load_async_to_lds_b128(GAS1(src), LAS3(nxt), 0, 0);
#else
            *(v8bf*)nxt = *(const v8bf*)src;
#endif
        }

        v16bf a = pack16(*(const v8bf*)(pa_base + k0),
                         *(const v8bf*)(pa_base + k0 + 16));
#pragma unroll
        for (int nt = 0; nt < 4; ++nt) {
            const __bf16* pb = cur + (nt * 16 + r) * WROW + half * 8;
            v16bf b = pack16(*(const v8bf*)pb, *(const v8bf*)(pb + 16));
            acc[nt] = wmma_bf16(a, b, acc[nt]);
        }
    }

#pragma unroll
    for (int nt = 0; nt < 4; ++nt) {
        int n = n0 + nt * 16 + r;
        float bv = bias[n];
#pragma unroll
        for (int j = 0; j < 8; ++j) {
            int m = m0 + j + half * 8;
            float val = acc[nt][j] + bv;
            if (MODE == 1) {
                xio[(size_t)m * E_ + n] += val;
            } else if (MODE == 2) {
                obf[(size_t)m * E_ + n] = (__bf16)fmaxf(val, 0.f);
            } else {
                int which = n >> 10;           // 0=q,1=k,2=v
                int f = n & 1023;
                int hh = f >> 6, d = f & 63;
                int sg = m >> 2, bq = m & 3;   // row m = s*B + b
                int bh = (bq << 4) | hh;
                if (which == 0)
                    qb[((size_t)bh * S_ + sg) * HD_ + d] =
                        (__bf16)(val * 0.00390625f);   // 1/(sqrt(hd)*sqrt(S))
                else if (which == 1)
                    kb[((size_t)bh * S_ + sg) * HD_ + d] = (__bf16)val;
                else
                    vtb[((size_t)bh * HD_ + d) * S_ + sg] = (__bf16)val;
            }
        }
    }
}

// ---------------------------------------------------------------------------
// Attention: grid (S/16, B*H), block 128 (4 waves).
// Phase1: logits (16 x 1024) via WMMA -> LDS fp32
// Phase2: softmax (wave-per-row, shfl reductions) -> LDS bf16 probs
// Phase3: probs x V^T via WMMA -> abuf (S*B, E) bf16
// ---------------------------------------------------------------------------
__global__ __launch_bounds__(128) void attn_kernel(
    const __bf16* __restrict__ qb, const __bf16* __restrict__ kb,
    const __bf16* __restrict__ vtb, __bf16* __restrict__ abuf) {
    __shared__ float  lg[16][1024];     // 64 KB
    __shared__ __bf16 pbuf[16][1024];   // 32 KB

    int lane = threadIdx.x & 31, wid = threadIdx.x >> 5;
    int half = lane >> 4, r = lane & 15;
    int s0 = blockIdx.x * 16;
    int bh = blockIdx.y;

    const __bf16* qh = qb + (size_t)bh * S_ * HD_;
    const __bf16* kh = kb + (size_t)bh * S_ * HD_;

    v16bf qa[2];
#pragma unroll
    for (int ds = 0; ds < 2; ++ds) {
        const __bf16* p = qh + (size_t)(s0 + r) * HD_ + ds * 32 + half * 8;
        qa[ds] = pack16(*(const v8bf*)p, *(const v8bf*)(p + 16));
    }

    for (int nt = wid * 16; nt < wid * 16 + 16; ++nt) {
        int t0 = nt * 16;
        v8f acc;
#pragma unroll
        for (int j = 0; j < 8; ++j) acc[j] = 0.f;
#pragma unroll
        for (int ds = 0; ds < 2; ++ds) {
            const __bf16* p = kh + (size_t)(t0 + r) * HD_ + ds * 32 + half * 8;
            v16bf b = pack16(*(const v8bf*)p, *(const v8bf*)(p + 16));
            acc = wmma_bf16(qa[ds], b, acc);
        }
#pragma unroll
        for (int j = 0; j < 8; ++j) lg[j + half * 8][t0 + r] = acc[j];
    }
    __syncthreads();

    for (int rr = wid; rr < 16; rr += 4) {
        float* Lr = lg[rr];
        float mx = -1e30f;
        for (int c = lane; c < 1024; c += 32) mx = fmaxf(mx, Lr[c]);
#pragma unroll
        for (int m = 16; m > 0; m >>= 1) mx = fmaxf(mx, __shfl_xor(mx, m, 32));
        float sm = 0.f;
        for (int c = lane; c < 1024; c += 32) {
            float e = __expf(Lr[c] - mx);
            Lr[c] = e;
            sm += e;
        }
#pragma unroll
        for (int m = 16; m > 0; m >>= 1) sm += __shfl_xor(sm, m, 32);
        float inv = 1.0f / sm;
        for (int c = lane; c < 1024; c += 32)
            pbuf[rr][c] = (__bf16)(Lr[c] * inv);
    }
    __syncthreads();

    int dbase = wid * 16;
    v8f acc;
#pragma unroll
    for (int j = 0; j < 8; ++j) acc[j] = 0.f;
    const __bf16* vh = vtb + ((size_t)bh * HD_ + dbase + r) * S_ + half * 8;
    for (int t0 = 0; t0 < 1024; t0 += 32) {
        const __bf16* pp = &pbuf[r][t0 + half * 8];
        v16bf a = pack16(*(const v8bf*)pp, *(const v8bf*)(pp + 16));
        v16bf b = pack16(*(const v8bf*)(vh + t0), *(const v8bf*)(vh + t0 + 16));
        acc = wmma_bf16(a, b, acc);
    }
    int bq = bh >> 4, hh = bh & 15;
#pragma unroll
    for (int j = 0; j < 8; ++j) {
        int sg = s0 + j + half * 8;
        int e = hh * HD_ + dbase + r;
        abuf[((size_t)sg * B_ + bq) * E_ + e] = (__bf16)acc[j];
    }
}

// ---------------------------------------------------------------------------
// Host orchestration
// ---------------------------------------------------------------------------
extern "C" void kernel_launch(void* const* d_in, const int* in_sizes, int n_in,
                              void* d_out, int out_size, void* d_ws, size_t ws_size,
                              hipStream_t stream) {
    const float* x_in  = (const float*)d_in[0];
    const float* wqkv  = (const float*)d_in[1];
    const float* bqkv  = (const float*)d_in[2];
    const float* wo    = (const float*)d_in[3];
    const float* bo    = (const float*)d_in[4];
    const float* w1    = (const float*)d_in[5];
    const float* b1    = (const float*)d_in[6];
    const float* w2    = (const float*)d_in[7];
    const float* b2    = (const float*)d_in[8];
    const float* ln0g  = (const float*)d_in[9];
    const float* ln0b  = (const float*)d_in[10];
    const float* ln1g  = (const float*)d_in[11];
    const float* ln1b  = (const float*)d_in[12];
    const float* lnfg  = (const float*)d_in[13];
    const float* lnfb  = (const float*)d_in[14];

    char* ws = (char*)d_ws;
    float*  xbuf = (float*)ws;            ws += (size_t)M_ * E_ * 4;        // 16 MB
    __bf16* hbf  = (__bf16*)ws;           ws += (size_t)M_ * E_ * 2;        // 8 MB
    __bf16* h1bf = (__bf16*)ws;           ws += (size_t)M_ * E_ * 2;        // 8 MB
    __bf16* abuf = (__bf16*)ws;           ws += (size_t)M_ * E_ * 2;        // 8 MB
    __bf16* qbuf = (__bf16*)ws;           ws += (size_t)64 * S_ * HD_ * 2;  // 8 MB
    __bf16* kbuf = (__bf16*)ws;           ws += (size_t)64 * S_ * HD_ * 2;  // 8 MB
    __bf16* vtbf = (__bf16*)ws;           ws += (size_t)64 * S_ * HD_ * 2;  // 8 MB
    __bf16* wqkvb = (__bf16*)ws;          ws += (size_t)L_ * 3 * E_ * E_ * 2;
    __bf16* wob   = (__bf16*)ws;          ws += (size_t)L_ * E_ * E_ * 2;
    __bf16* w1b   = (__bf16*)ws;          ws += (size_t)L_ * E_ * E_ * 2;
    __bf16* w2b   = (__bf16*)ws;          ws += (size_t)L_ * E_ * E_ * 2;

    {
        int n = L_ * 3 * E_ * E_;
        cvt_bf16_kernel<<<n / 256, 256, 0, stream>>>(wqkv, wqkvb, n);
        n = L_ * E_ * E_;
        cvt_bf16_kernel<<<n / 256, 256, 0, stream>>>(wo, wob, n);
        cvt_bf16_kernel<<<n / 256, 256, 0, stream>>>(w1, w1b, n);
        cvt_bf16_kernel<<<n / 256, 256, 0, stream>>>(w2, w2b, n);
    }

    posadd_kernel<<<(M_ * E_) / 256, 256, 0, stream>>>(x_in, xbuf);

    dim3 gQKV(3 * E_ / 64, M_ / 128);   // 48 x 32
    dim3 gSQ(E_ / 64, M_ / 128);        // 16 x 32
    dim3 gAT(S_ / 16, B_ * H_);         // 64 x 64

    for (int l = 0; l < L_; ++l) {
        const __bf16* wqkv_l = wqkvb + (size_t)l * 3 * E_ * E_;
        const __bf16* wo_l   = wob   + (size_t)l * E_ * E_;
        const __bf16* w1_l   = w1b   + (size_t)l * E_ * E_;
        const __bf16* w2_l   = w2b   + (size_t)l * E_ * E_;

        ln_kernel<1><<<M_, 256, 0, stream>>>(xbuf, ln0g + l * E_, ln0b + l * E_,
                                             hbf, nullptr);
        gemm_bf16_kernel<0><<<gQKV, 256, 0, stream>>>(
            hbf, wqkv_l, bqkv + (size_t)l * 3 * E_, nullptr, nullptr,
            qbuf, kbuf, vtbf);
        attn_kernel<<<gAT, 128, 0, stream>>>(qbuf, kbuf, vtbf, abuf);
        gemm_bf16_kernel<1><<<gSQ, 256, 0, stream>>>(
            abuf, wo_l, bo + (size_t)l * E_, xbuf, nullptr,
            nullptr, nullptr, nullptr);

        ln_kernel<1><<<M_, 256, 0, stream>>>(xbuf, ln1g + l * E_, ln1b + l * E_,
                                             hbf, nullptr);
        gemm_bf16_kernel<2><<<gSQ, 256, 0, stream>>>(
            hbf, w1_l, b1 + (size_t)l * E_, nullptr, h1bf,
            nullptr, nullptr, nullptr);
        gemm_bf16_kernel<1><<<gSQ, 256, 0, stream>>>(
            h1bf, w2_l, b2 + (size_t)l * E_, xbuf, nullptr,
            nullptr, nullptr, nullptr);
    }

    ln_kernel<0><<<M_, 256, 0, stream>>>(xbuf, lnfg, lnfb, nullptr,
                                         (float*)d_out);
}
